// DecoderDGLConcat_42777874268716
// MI455X (gfx1250) — matrix operand order repaired and checked by reading, not compiled, not run on hardware
//
#include <hip/hip_runtime.h>
#include <hip/hip_bf16.h>

typedef __attribute__((ext_vector_type(16))) __bf16 v16bf;
typedef __attribute__((ext_vector_type(8)))  float  v8f;
typedef __attribute__((ext_vector_type(4)))  float  f4v;

struct alignas(16) U4 { unsigned int a, b, c, d; };
struct alignas(8)  U2 { unsigned int a, b; };
union Frag { v16bf v; U4 q[2]; };

#define DD 512

__device__ __forceinline__ unsigned short f2bf(float f) {
    unsigned int u = __float_as_uint(f);
    u += 0x7FFFu + ((u >> 16) & 1u);          // round-to-nearest-even
    return (unsigned short)(u >> 16);
}

__device__ __forceinline__ void st4bf(unsigned short* d, f4v v) {
    U2 u;
    u.a = (unsigned int)f2bf(v[0]) | ((unsigned int)f2bf(v[1]) << 16);
    u.b = (unsigned int)f2bf(v[2]) | ((unsigned int)f2bf(v[3]) << 16);
    *(U2*)d = u;
}

__device__ __forceinline__ float red16(float v) {
    #pragma unroll
    for (int m = 1; m < 16; m <<= 1) v += __shfl_xor(v, m, 32);
    return v;
}

// ---------------------------------------------------------------- utilities
__global__ void zero_f32_kernel(float* p, long n) {
    long i = (long)blockIdx.x * blockDim.x + threadIdx.x;
    long stride = (long)gridDim.x * blockDim.x;
    for (; i < n; i += stride) p[i] = 0.0f;
}

// in: [K][N] row-major fp32  ->  out: [N][K] bf16 (transposed)
__global__ void cvt_w_kernel(const float* __restrict__ in,
                             unsigned short* __restrict__ out, int K, int N) {
    long idx = (long)blockIdx.x * blockDim.x + threadIdx.x;
    long total = (long)K * N;
    if (idx >= total) return;
    int k = (int)(idx / N), n = (int)(idx % N);
    out[(long)n * K + k] = f2bf(in[idx]);
}

// ---------------------------------------------------------------- edge MLP
// 32 edges per block, 256 threads (8 waves). wave w owns cols [64w, 64w+64)
// and BOTH 16-row M-tiles: each B fragment is loaded once, used twice.
__global__ __launch_bounds__(256)
void edge_mlp_kernel(const float* __restrict__ m2g, const float* __restrict__ mesh,
                     const float* __restrict__ grd, const int* __restrict__ src,
                     const int* __restrict__ dst,
                     const unsigned short* __restrict__ w1t, const float* __restrict__ b1,
                     const unsigned short* __restrict__ w2t, const float* __restrict__ b2,
                     const float* __restrict__ lng, const float* __restrict__ lnb,
                     float* __restrict__ agg)
{
    constexpr int K1 = 3 * DD;           // 1536
    constexpr int XS = K1 + 8;           // LDS row stride (pad -> 4-bank skew)
    constexpr int HS = DD + 8;
    extern __shared__ char smem_raw[];
    float* s1m = (float*)smem_raw;                         // 32 floats
    float* s2m = (float*)(smem_raw + 128);                 // 32 floats
    int*   dsti = (int*)(smem_raw + 256);                  // 32 ints
    unsigned short* xs = (unsigned short*)(smem_raw + 384);// 32*XS bf16 (98.8 KB)
    unsigned short* hs = xs;                               // overlay after GEMM1

    const int tid  = threadIdx.x;
    const int lane = tid & 31, wave = tid >> 5;
    const int l15  = lane & 15, hi = lane >> 4;
    const long base = (long)blockIdx.x * 32;

    if (tid < 32) dsti[tid] = dst[base + tid];

    // ---- stage concat([efeat | mesh[src] | grid[dst]]) as bf16, 4 rows/wave.
    //      m2g is a no-reuse 402MB stream -> non-temporal to spare L2.
    #pragma unroll
    for (int rr = 0; rr < 4; ++rr) {
        int r = wave * 4 + rr;
        long e = base + r;
        const f4v* p0 = (const f4v*)(m2g + e * DD);
        const f4v* p1 = (const f4v*)(mesh + (long)src[e] * DD);
        const f4v* p2 = (const f4v*)(grd  + (long)dst[e] * DD);
        unsigned short* xr = xs + r * XS;
        for (int t = lane; t < DD / 4; t += 32) {
            st4bf(xr + 4 * t,          __builtin_nontemporal_load(p0 + t));
            st4bf(xr + DD + 4 * t,     p1[t]);
            st4bf(xr + 2 * DD + 4 * t, p2[t]);
        }
    }
    __syncthreads();

    // ---- GEMM1: h = X(32x1536) * W1(1536x512), wave slice = 64 cols x 32 rows
    const int colbase = wave * 64;
    v8f acc[2][4] = {};
    const unsigned short* xr0 = xs + l15 * XS;
    const unsigned short* xr1 = xs + (16 + l15) * XS;
    for (int k = 0; k < K1; k += 32) {
        Frag a0, a1;
        a0.q[0] = *(const U4*)(xr0 + k + hi * 8);
        a0.q[1] = *(const U4*)(xr0 + k + 16 + hi * 8);
        a1.q[0] = *(const U4*)(xr1 + k + hi * 8);
        a1.q[1] = *(const U4*)(xr1 + k + 16 + hi * 8);
        #pragma unroll
        for (int nt = 0; nt < 4; ++nt) {
            const unsigned short* wr =
                w1t + (long)(colbase + nt * 16 + l15) * K1 + k + hi * 8;
            Frag b;
            b.q[0] = ((const U4*)wr)[0];
            b.q[1] = ((const U4*)wr)[2];
            acc[0][nt] = __builtin_amdgcn_wmma_f32_16x16x32_bf16(
                false, a0.v, false, b.v, (short)0, acc[0][nt], false, false);
            acc[1][nt] = __builtin_amdgcn_wmma_f32_16x16x32_bf16(
                false, a1.v, false, b.v, (short)0, acc[1][nt], false, false);
        }
    }
    __syncthreads();   // everyone done reading xs

    // ---- bias + SiLU -> hs (bf16)
    #pragma unroll
    for (int mt = 0; mt < 2; ++mt)
        #pragma unroll
        for (int nt = 0; nt < 4; ++nt) {
            int col = colbase + nt * 16 + l15;
            float bv = b1[col];
            #pragma unroll
            for (int i = 0; i < 8; ++i) {
                float h = acc[mt][nt][i] + bv;
                float s = h * (1.0f / (1.0f + __expf(-h)));
                hs[(mt * 16 + i + hi * 8) * HS + col] = f2bf(s);
            }
        }
    __syncthreads();

    // ---- GEMM2: y = h(32x512) * W2(512x512)
    v8f acc2[2][4] = {};
    const unsigned short* hr0 = hs + l15 * HS;
    const unsigned short* hr1 = hs + (16 + l15) * HS;
    for (int k = 0; k < DD; k += 32) {
        Frag a0, a1;
        a0.q[0] = *(const U4*)(hr0 + k + hi * 8);
        a0.q[1] = *(const U4*)(hr0 + k + 16 + hi * 8);
        a1.q[0] = *(const U4*)(hr1 + k + hi * 8);
        a1.q[1] = *(const U4*)(hr1 + k + 16 + hi * 8);
        #pragma unroll
        for (int nt = 0; nt < 4; ++nt) {
            const unsigned short* wr =
                w2t + (long)(colbase + nt * 16 + l15) * DD + k + hi * 8;
            Frag b;
            b.q[0] = ((const U4*)wr)[0];
            b.q[1] = ((const U4*)wr)[2];
            acc2[0][nt] = __builtin_amdgcn_wmma_f32_16x16x32_bf16(
                false, a0.v, false, b.v, (short)0, acc2[0][nt], false, false);
            acc2[1][nt] = __builtin_amdgcn_wmma_f32_16x16x32_bf16(
                false, a1.v, false, b.v, (short)0, acc2[1][nt], false, false);
        }
    }
    #pragma unroll
    for (int mt = 0; mt < 2; ++mt)
        #pragma unroll
        for (int nt = 0; nt < 4; ++nt) {
            float bv = b2[colbase + nt * 16 + l15];
            #pragma unroll
            for (int i = 0; i < 8; ++i) acc2[mt][nt][i] += bv;
        }

    // ---- LayerNorm stats across 512 cols per row
    if (tid < 32) { s1m[tid] = 0.0f; s2m[tid] = 0.0f; }
    __syncthreads();
    float ps[2][8] = {}, pq[2][8] = {};
    #pragma unroll
    for (int mt = 0; mt < 2; ++mt)
        #pragma unroll
        for (int nt = 0; nt < 4; ++nt)
            #pragma unroll
            for (int i = 0; i < 8; ++i) {
                float v = acc2[mt][nt][i];
                ps[mt][i] += v; pq[mt][i] += v * v;
            }
    #pragma unroll
    for (int mt = 0; mt < 2; ++mt)
        #pragma unroll
        for (int i = 0; i < 8; ++i) {
            ps[mt][i] = red16(ps[mt][i]);
            pq[mt][i] = red16(pq[mt][i]);
        }
    if (l15 == 0) {
        #pragma unroll
        for (int mt = 0; mt < 2; ++mt)
            #pragma unroll
            for (int i = 0; i < 8; ++i) {
                int row = mt * 16 + i + hi * 8;
                atomicAdd(&s1m[row], ps[mt][i]);
                atomicAdd(&s2m[row], pq[mt][i]);
            }
    }
    __syncthreads();

    float mean[2][8], rstd[2][8];
    #pragma unroll
    for (int mt = 0; mt < 2; ++mt)
        #pragma unroll
        for (int i = 0; i < 8; ++i) {
            int row = mt * 16 + i + hi * 8;
            float m = s1m[row] * (1.0f / DD);
            float v = s2m[row] * (1.0f / DD) - m * m;
            mean[mt][i] = m; rstd[mt][i] = rsqrtf(v + 1e-5f);
        }

    // ---- normalize + fused scatter-add into agg[dst]
    #pragma unroll
    for (int mt = 0; mt < 2; ++mt)
        #pragma unroll
        for (int nt = 0; nt < 4; ++nt) {
            int col = colbase + nt * 16 + l15;
            float g = lng[col], bb = lnb[col];
            #pragma unroll
            for (int i = 0; i < 8; ++i) {
                int row = mt * 16 + i + hi * 8;
                float y = (acc2[mt][nt][i] - mean[mt][i]) * rstd[mt][i] * g + bb;
                atomicAdd(agg + (long)dsti[row] * DD + col, y);
            }
        }
}

// ---------------------------------------------------------------- node MLP
// 32 grid nodes per block, same 2-Mtile B-reuse structure.
__global__ __launch_bounds__(256)
void node_mlp_kernel(const float* __restrict__ agg, const float* __restrict__ grd,
                     const unsigned short* __restrict__ w1t, const float* __restrict__ b1,
                     const unsigned short* __restrict__ w2t, const float* __restrict__ b2,
                     const float* __restrict__ lng, const float* __restrict__ lnb,
                     float* __restrict__ out)
{
    constexpr int K1 = 2 * DD;           // 1024
    constexpr int XS = K1 + 8;
    constexpr int HS = DD + 8;
    extern __shared__ char smem_raw[];
    float* s1m = (float*)smem_raw;
    float* s2m = (float*)(smem_raw + 128);
    unsigned short* xs = (unsigned short*)(smem_raw + 256); // 32*XS bf16 (66 KB)
    unsigned short* hs = xs;

    const int tid  = threadIdx.x;
    const int lane = tid & 31, wave = tid >> 5;
    const int l15  = lane & 15, hi = lane >> 4;
    const long base = (long)blockIdx.x * 32;

    #pragma unroll
    for (int rr = 0; rr < 4; ++rr) {
        int r = wave * 4 + rr;
        long n = base + r;
        const f4v* p0 = (const f4v*)(agg + n * DD);
        const f4v* p1 = (const f4v*)(grd + n * DD);
        unsigned short* xr = xs + r * XS;
        for (int t = lane; t < DD / 4; t += 32) {
            st4bf(xr + 4 * t,      p0[t]);
            st4bf(xr + DD + 4 * t, p1[t]);
        }
    }
    __syncthreads();

    const int colbase = wave * 64;
    v8f acc[2][4] = {};
    const unsigned short* xr0 = xs + l15 * XS;
    const unsigned short* xr1 = xs + (16 + l15) * XS;
    for (int k = 0; k < K1; k += 32) {
        Frag a0, a1;
        a0.q[0] = *(const U4*)(xr0 + k + hi * 8);
        a0.q[1] = *(const U4*)(xr0 + k + 16 + hi * 8);
        a1.q[0] = *(const U4*)(xr1 + k + hi * 8);
        a1.q[1] = *(const U4*)(xr1 + k + 16 + hi * 8);
        #pragma unroll
        for (int nt = 0; nt < 4; ++nt) {
            const unsigned short* wr =
                w1t + (long)(colbase + nt * 16 + l15) * K1 + k + hi * 8;
            Frag b;
            b.q[0] = ((const U4*)wr)[0];
            b.q[1] = ((const U4*)wr)[2];
            acc[0][nt] = __builtin_amdgcn_wmma_f32_16x16x32_bf16(
                false, a0.v, false, b.v, (short)0, acc[0][nt], false, false);
            acc[1][nt] = __builtin_amdgcn_wmma_f32_16x16x32_bf16(
                false, a1.v, false, b.v, (short)0, acc[1][nt], false, false);
        }
    }
    __syncthreads();

    #pragma unroll
    for (int mt = 0; mt < 2; ++mt)
        #pragma unroll
        for (int nt = 0; nt < 4; ++nt) {
            int col = colbase + nt * 16 + l15;
            float bv = b1[col];
            #pragma unroll
            for (int i = 0; i < 8; ++i) {
                float h = acc[mt][nt][i] + bv;
                float s = h * (1.0f / (1.0f + __expf(-h)));
                hs[(mt * 16 + i + hi * 8) * HS + col] = f2bf(s);
            }
        }
    __syncthreads();

    v8f acc2[2][4] = {};
    const unsigned short* hr0 = hs + l15 * HS;
    const unsigned short* hr1 = hs + (16 + l15) * HS;
    for (int k = 0; k < DD; k += 32) {
        Frag a0, a1;
        a0.q[0] = *(const U4*)(hr0 + k + hi * 8);
        a0.q[1] = *(const U4*)(hr0 + k + 16 + hi * 8);
        a1.q[0] = *(const U4*)(hr1 + k + hi * 8);
        a1.q[1] = *(const U4*)(hr1 + k + 16 + hi * 8);
        #pragma unroll
        for (int nt = 0; nt < 4; ++nt) {
            const unsigned short* wr =
                w2t + (long)(colbase + nt * 16 + l15) * DD + k + hi * 8;
            Frag b;
            b.q[0] = ((const U4*)wr)[0];
            b.q[1] = ((const U4*)wr)[2];
            acc2[0][nt] = __builtin_amdgcn_wmma_f32_16x16x32_bf16(
                false, a0.v, false, b.v, (short)0, acc2[0][nt], false, false);
            acc2[1][nt] = __builtin_amdgcn_wmma_f32_16x16x32_bf16(
                false, a1.v, false, b.v, (short)0, acc2[1][nt], false, false);
        }
    }
    #pragma unroll
    for (int mt = 0; mt < 2; ++mt)
        #pragma unroll
        for (int nt = 0; nt < 4; ++nt) {
            float bv = b2[colbase + nt * 16 + l15];
            #pragma unroll
            for (int i = 0; i < 8; ++i) acc2[mt][nt][i] += bv;
        }

    if (tid < 32) { s1m[tid] = 0.0f; s2m[tid] = 0.0f; }
    __syncthreads();
    float ps[2][8] = {}, pq[2][8] = {};
    #pragma unroll
    for (int mt = 0; mt < 2; ++mt)
        #pragma unroll
        for (int nt = 0; nt < 4; ++nt)
            #pragma unroll
            for (int i = 0; i < 8; ++i) {
                float v = acc2[mt][nt][i];
                ps[mt][i] += v; pq[mt][i] += v * v;
            }
    #pragma unroll
    for (int mt = 0; mt < 2; ++mt)
        #pragma unroll
        for (int i = 0; i < 8; ++i) {
            ps[mt][i] = red16(ps[mt][i]);
            pq[mt][i] = red16(pq[mt][i]);
        }
    if (l15 == 0) {
        #pragma unroll
        for (int mt = 0; mt < 2; ++mt)
            #pragma unroll
            for (int i = 0; i < 8; ++i) {
                int row = mt * 16 + i + hi * 8;
                atomicAdd(&s1m[row], ps[mt][i]);
                atomicAdd(&s2m[row], pq[mt][i]);
            }
    }
    __syncthreads();

    float mean[2][8], rstd[2][8];
    #pragma unroll
    for (int mt = 0; mt < 2; ++mt)
        #pragma unroll
        for (int i = 0; i < 8; ++i) {
            int row = mt * 16 + i + hi * 8;
            float m = s1m[row] * (1.0f / DD);
            float v = s2m[row] * (1.0f / DD) - m * m;
            mean[mt][i] = m; rstd[mt][i] = rsqrtf(v + 1e-5f);
        }

    #pragma unroll
    for (int mt = 0; mt < 2; ++mt)
        #pragma unroll
        for (int nt = 0; nt < 4; ++nt) {
            int col = colbase + nt * 16 + l15;
            float g = lng[col], bb = lnb[col];
            #pragma unroll
            for (int i = 0; i < 8; ++i) {
                long n = base + mt * 16 + i + hi * 8;
                float y = (acc2[mt][nt][i] - mean[mt][i]) * rstd[mt][i] * g + bb
                        + grd[n * DD + col];
                __builtin_nontemporal_store(y, &out[n * DD + col]);
            }
        }
}

// ---------------------------------------------------------------- launch
extern "C" void kernel_launch(void* const* d_in, const int* in_sizes, int n_in,
                              void* d_out, int out_size, void* d_ws, size_t ws_size,
                              hipStream_t stream) {
    (void)in_sizes; (void)n_in; (void)out_size; (void)ws_size;
    const int N_GRID = 65536, N_EDGE = 196608;

    const float* m2g  = (const float*)d_in[0];
    const float* grd  = (const float*)d_in[1];
    const float* mesh = (const float*)d_in[2];
    const float* eW1  = (const float*)d_in[3];
    const float* eb1  = (const float*)d_in[4];
    const float* eW2  = (const float*)d_in[5];
    const float* eb2  = (const float*)d_in[6];
    const float* eg   = (const float*)d_in[7];
    const float* ebn  = (const float*)d_in[8];
    const float* nW1  = (const float*)d_in[9];
    const float* nb1  = (const float*)d_in[10];
    const float* nW2  = (const float*)d_in[11];
    const float* nb2  = (const float*)d_in[12];
    const float* ng   = (const float*)d_in[13];
    const float* nbn  = (const float*)d_in[14];
    const int*   src  = (const int*)d_in[15];
    const int*   dst  = (const int*)d_in[16];
    float* out = (float*)d_out;

    char* ws = (char*)d_ws;
    size_t off = 0;
    float* agg = (float*)(ws + off);                   off += (size_t)N_GRID * DD * 4;
    unsigned short* w1e = (unsigned short*)(ws + off); off += (size_t)DD * 3 * DD * 2;
    unsigned short* w2e = (unsigned short*)(ws + off); off += (size_t)DD * DD * 2;
    unsigned short* w1n = (unsigned short*)(ws + off); off += (size_t)DD * 2 * DD * 2;
    unsigned short* w2n = (unsigned short*)(ws + off); off += (size_t)DD * DD * 2;

    zero_f32_kernel<<<4096, 256, 0, stream>>>(agg, (long)N_GRID * DD);
    cvt_w_kernel<<<(3 * DD * DD + 255) / 256, 256, 0, stream>>>(eW1, w1e, 3 * DD, DD);
    cvt_w_kernel<<<(DD * DD + 255) / 256, 256, 0, stream>>>(eW2, w2e, DD, DD);
    cvt_w_kernel<<<(2 * DD * DD + 255) / 256, 256, 0, stream>>>(nW1, w1n, 2 * DD, DD);
    cvt_w_kernel<<<(DD * DD + 255) / 256, 256, 0, stream>>>(nW2, w2n, DD, DD);

    const size_t edge_lds = 384 + (size_t)32 * (3 * DD + 8) * 2;   // ~99.2 KB
    const size_t node_lds = 256 + (size_t)32 * (2 * DD + 8) * 2;   // ~66.3 KB
    edge_mlp_kernel<<<N_EDGE / 32, 256, edge_lds, stream>>>(
        m2g, mesh, grd, src, dst, w1e, eb1, w2e, eb2, eg, ebn, agg);
    node_mlp_kernel<<<N_GRID / 32, 256, node_lds, stream>>>(
        agg, grd, w1n, nb1, w2n, nb2, ng, nbn, out);
}
